// InformerEncoderLayer_51196010168630
// MI455X (gfx1250) — matrix operand order repaired and checked by reading, not compiled, not run on hardware
//
#include <hip/hip_runtime.h>
#include <hip/hip_bf16.h>
#include <math.h>

// ---------------- problem constants (from reference) ----------------
#define B_   4
#define L_   2048
#define DM_  512
#define H_   8
#define DFF_ 2048
#define D_   64          // DM / H
#define U_   40          // min(5*ceil(ln(2048)), 2048) = 5*8
#define EPS_ 1e-5f
#define KC_  64          // K-chunk staged in LDS per double-buffer step

typedef __attribute__((ext_vector_type(16))) _Float16 v16h;
typedef __attribute__((ext_vector_type(8)))  _Float16 v8h;
typedef __attribute__((ext_vector_type(8)))  float    v8f;
typedef int v4i_async __attribute__((__vector_size__(16)));

#ifndef __has_builtin
#define __has_builtin(x) 0
#endif
#if defined(__AMDGCN__) && __has_builtin(__builtin_amdgcn_global_load_async_to_lds_b128)
#define USE_ASYNC_LDS 1
// arg0: AS(1) v4i* (global src), arg1: AS(3) v4i* (LDS dst), arg2: imm offset, arg3: cpol
#define ASYNC_COPY_B128(g, l)                                          \
  __builtin_amdgcn_global_load_async_to_lds_b128(                      \
      (__attribute__((address_space(1))) v4i_async*)(g),               \
      (__attribute__((address_space(3))) v4i_async*)(l), 0, 0)
#define ASYNC_WAIT_2() asm volatile("s_wait_asynccnt 0x2" ::: "memory")
#define ASYNC_WAIT_0() asm volatile("s_wait_asynccnt 0x0" ::: "memory")
#else
#define USE_ASYNC_LDS 0
#define ASYNC_WAIT_2()
#define ASYNC_WAIT_0()
#endif

// ---------------- f32 -> f16 convert ----------------
__global__ __launch_bounds__(256) void cvt_f16_kernel(const float* __restrict__ a,
                                                      _Float16* __restrict__ o, int n) {
  int t = blockIdx.x * 256 + threadIdx.x;
  if (t < n) o[t] = (_Float16)a[t];
}

// W is (Kd, Nd) row-major f32 -> Wt (Nd, Kd) row-major f16
__global__ __launch_bounds__(256) void transpose_cvt_kernel(const float* __restrict__ W,
                                                            _Float16* __restrict__ Wt,
                                                            int Kd, int Nd) {
  int t = blockIdx.x * 256 + threadIdx.x;
  if (t >= Kd * Nd) return;
  int n = t % Nd, k = t / Nd;
  Wt[(size_t)n * Kd + k] = (_Float16)W[t];
}

// ---------------- WMMA GEMM: C[M,N] = A[M,K] * Bt[N,K]^T + bias ----------------
// Block = 4 waves. Block tile: 64 (M) x 32 (N). Each wave: 16 (M) x 32 (N),
// i.e. two 16x16 WMMA accumulators sharing one A fragment per K step.
// B tile (32 rows of Bt x KC halves) is staged in LDS, double-buffered, via
// async global->LDS copies (all 4 waves share it).
#define BS_STRIDE 72   // padded LDS row stride in halves (144B, multiple of 16B)
template<int RELU>
__global__ __launch_bounds__(128) void gemm_f16_wmma_kernel(
    const _Float16* __restrict__ A,   // M x K row-major f16
    const _Float16* __restrict__ Bt,  // N x K row-major f16
    const float*    __restrict__ bias,// N (may be null)
    float*          __restrict__ Cf,  // M x N f32 out (may be null)
    _Float16*       __restrict__ Ch,  // M x N f16 out (may be null)
    int M, int N, int K)
{
  __shared__ _Float16 Bs[2][32 * BS_STRIDE];

  const int tid  = threadIdx.x;
  const int wave = tid >> 5;
  const int lane = tid & 31;
  const int half = lane >> 4;     // 0: lanes 0-15, 1: lanes 16-31
  const int r    = lane & 15;

  const int n0 = blockIdx.x * 32;
  const int m0 = (blockIdx.y * 4 + wave) * 16;

  const _Float16* arow = A + (size_t)(m0 + r) * K;

  // B staging map: 32 rows x KC(=64) halves = 4KB = 256 x 16B transfers,
  // 128 threads -> 2 transfers each (rows t>>3 and (t>>3)+16, segment t&7).
  const int row0 = tid >> 3, seg = tid & 7;
  const int row1 = row0 + 16;
  const _Float16* bg0 = Bt + (size_t)(n0 + row0) * K + seg * 8;
  const _Float16* bg1 = Bt + (size_t)(n0 + row1) * K + seg * 8;
  _Float16* bl0base = &Bs[0][row0 * BS_STRIDE + seg * 8];
  _Float16* bl1base = &Bs[0][row1 * BS_STRIDE + seg * 8];
  const int bufOfs = 32 * BS_STRIDE;

  auto stage = [&](int c, int buf) {
    const _Float16* g0 = bg0 + c * KC_;
    const _Float16* g1 = bg1 + c * KC_;
    _Float16* l0 = bl0base + buf * bufOfs;
    _Float16* l1 = bl1base + buf * bufOfs;
#if USE_ASYNC_LDS
    ASYNC_COPY_B128(g0, l0);
    ASYNC_COPY_B128(g1, l1);
#else
    *(v8h*)l0 = *(const v8h*)g0;
    *(v8h*)l1 = *(const v8h*)g1;
#endif
  };

  const int nch = K / KC_;
  stage(0, 0);

  v8f acc0 = {}, acc1 = {};
  for (int c = 0; c < nch; ++c) {
    const int buf = c & 1;
    if (c + 1 < nch) {
      __syncthreads();              // prior reads of buf^1 complete
      stage(c + 1, buf ^ 1);        // prefetch next chunk
      ASYNC_WAIT_2();               // wait for chunk c only (in-order)
    } else {
      ASYNC_WAIT_0();
    }
    __syncthreads();                // chunk c visible to all waves

    const int kbase = c * KC_;
    if (kbase + KC_ < K) {
      __builtin_prefetch(arow + kbase + KC_, 0, 3);
    }
#pragma unroll
    for (int ko = 0; ko < KC_; ko += 32) {
      // A fragment (16x32): lane<16 holds K +0..15, lane>=16 K +16..31
      v8h a_lo = *(const v8h*)(arow + kbase + ko + half * 16);
      v8h a_hi = *(const v8h*)(arow + kbase + ko + half * 16 + 8);
      // B fragments from LDS: lane<16 -> K +0..15, lane>=16 -> K +16..31
      const _Float16* bp0 = &Bs[buf][r * BS_STRIDE + ko + half * 16];
      const _Float16* bp1 = &Bs[buf][(r + 16) * BS_STRIDE + ko + half * 16];
      v8h b0l = *(const v8h*)bp0, b0h = *(const v8h*)(bp0 + 8);
      v8h b1l = *(const v8h*)bp1, b1h = *(const v8h*)(bp1 + 8);
      v16h a, b0, b1;
#pragma unroll
      for (int i = 0; i < 8; ++i) {
        a[i]  = a_lo[i]; a[i + 8]  = a_hi[i];
        b0[i] = b0l[i];  b0[i + 8] = b0h[i];
        b1[i] = b1l[i];  b1[i + 8] = b1h[i];
      }
      acc0 = __builtin_amdgcn_wmma_f32_16x16x32_f16(false, a, false, b0,
                                                    (short)0, acc0, false, false);
      acc1 = __builtin_amdgcn_wmma_f32_16x16x32_f16(false, a, false, b1,
                                                    (short)0, acc1, false, false);
    }
  }

  // C/D layout: acc[i] -> row m0 + i + 8*half, col n0 + r (acc1: +16)
  const int cn0 = n0 + r;
  const int cn1 = n0 + 16 + r;
  const float bv0 = bias ? bias[cn0] : 0.0f;
  const float bv1 = bias ? bias[cn1] : 0.0f;
#pragma unroll
  for (int i = 0; i < 8; ++i) {
    int cm = m0 + i + 8 * half;
    float v0 = acc0[i] + bv0;
    float v1 = acc1[i] + bv1;
    if (RELU) { v0 = v0 > 0.0f ? v0 : 0.0f; v1 = v1 > 0.0f ? v1 : 0.0f; }
    if (Cf) {
      Cf[(size_t)cm * N + cn0] = v0;
      Cf[(size_t)cm * N + cn1] = v1;
    }
    if (Ch) {
      Ch[(size_t)cm * N + cn0] = (_Float16)v0;
      Ch[(size_t)cm * N + cn1] = (_Float16)v1;
    }
  }
}

// ---------------- sampled QK scores: Mscore[b,h,l] = max_u - mean_u ----------------
__global__ __launch_bounds__(256) void sampled_scores_kernel(
    const float* __restrict__ q, const float* __restrict__ k,
    const int* __restrict__ sidx, float* __restrict__ Mscore)
{
  int t = blockIdx.x * 256 + threadIdx.x;           // over B*H*L
  if (t >= B_ * H_ * L_) return;
  int l = t % L_;
  int h = (t / L_) % H_;
  int b = t / (L_ * H_);
  const float4* qp = (const float4*)(q + ((size_t)(b * L_ + l)) * DM_ + h * D_);
  float4 qr[16];
#pragma unroll
  for (int j = 0; j < 16; ++j) qr[j] = qp[j];
  float mx = -INFINITY, sum = 0.0f;
  for (int u = 0; u < U_; ++u) {
    int li = sidx[l * U_ + u];
    const float4* kp = (const float4*)(k + ((size_t)(b * L_ + li)) * DM_ + h * D_);
    float dot = 0.0f;
#pragma unroll
    for (int j = 0; j < 16; ++j) {
      float4 a = qr[j], c = kp[j];
      dot += a.x * c.x + a.y * c.y + a.z * c.z + a.w * c.w;
    }
    mx = fmaxf(mx, dot);
    sum += dot;
  }
  Mscore[t] = mx - sum * (1.0f / U_);
}

// ---------------- iterative top-U argmax per (b,h) ----------------
__global__ __launch_bounds__(256) void topk_kernel(const float* __restrict__ Mscore,
                                                   int* __restrict__ Mtop)
{
  __shared__ float vals[L_];
  __shared__ float rv[256];
  __shared__ int   ri[256];
  int bh = blockIdx.x;
  const float* src = Mscore + (size_t)bh * L_;
  for (int i = threadIdx.x; i < L_; i += 256) vals[i] = src[i];
  __syncthreads();
  for (int it = 0; it < U_; ++it) {
    float best = -INFINITY; int bi = 0;
    for (int i = threadIdx.x; i < L_; i += 256)
      if (vals[i] > best) { best = vals[i]; bi = i; }
    rv[threadIdx.x] = best; ri[threadIdx.x] = bi;
    __syncthreads();
    for (int s = 128; s > 0; s >>= 1) {
      if (threadIdx.x < s && rv[threadIdx.x + s] > rv[threadIdx.x]) {
        rv[threadIdx.x] = rv[threadIdx.x + s];
        ri[threadIdx.x] = ri[threadIdx.x + s];
      }
      __syncthreads();
    }
    if (threadIdx.x == 0) { Mtop[bh * U_ + it] = ri[0]; vals[ri[0]] = -INFINITY; }
    __syncthreads();
  }
}

// ---------------- softmax attention for selected queries ----------------
__global__ __launch_bounds__(256) void attn_selected_kernel(
    const float* __restrict__ q, const float* __restrict__ k,
    const float* __restrict__ v, const int* __restrict__ Mtop,
    float* __restrict__ ctx_top)
{
  __shared__ float p[L_];
  __shared__ float red[256];
  __shared__ float qs[D_];
  int u  = blockIdx.x % U_;
  int bh = blockIdx.x / U_;
  int h  = bh % H_;
  int b  = bh / H_;
  int lq = Mtop[bh * U_ + u];
  if (threadIdx.x < D_)
    qs[threadIdx.x] = q[((size_t)(b * L_ + lq)) * DM_ + h * D_ + threadIdx.x];
  __syncthreads();

  const float scale = 0.125f;   // 1/sqrt(64)
  float lmax = -INFINITY;
  for (int l = threadIdx.x; l < L_; l += 256) {
    const float* kp = k + ((size_t)(b * L_ + l)) * DM_ + h * D_;
    float dot = 0.0f;
#pragma unroll 8
    for (int d = 0; d < D_; ++d) dot += qs[d] * kp[d];
    dot *= scale;
    p[l] = dot;
    lmax = fmaxf(lmax, dot);
  }
  red[threadIdx.x] = lmax; __syncthreads();
  for (int s = 128; s > 0; s >>= 1) {
    if (threadIdx.x < s) red[threadIdx.x] = fmaxf(red[threadIdx.x], red[threadIdx.x + s]);
    __syncthreads();
  }
  float gmax = red[0]; __syncthreads();

  float lsum = 0.0f;
  for (int l = threadIdx.x; l < L_; l += 256) {
    float e = __expf(p[l] - gmax);
    p[l] = e;
    lsum += e;
  }
  red[threadIdx.x] = lsum; __syncthreads();
  for (int s = 128; s > 0; s >>= 1) {
    if (threadIdx.x < s) red[threadIdx.x] += red[threadIdx.x + s];
    __syncthreads();
  }
  float inv = 1.0f / red[0];
  __syncthreads();

  int d = threadIdx.x & 63, c = threadIdx.x >> 6;   // 4 chunks of 512 rows
  float acc = 0.0f;
  for (int l = c * 512; l < (c + 1) * 512; ++l)
    acc += p[l] * v[((size_t)(b * L_ + l)) * DM_ + h * D_ + d];
  red[threadIdx.x] = acc; __syncthreads();
  if (c == 0) {
    float s = red[d] + red[d + 64] + red[d + 128] + red[d + 192];
    ctx_top[((size_t)bh * U_ + u) * D_ + d] = s * inv;
  }
}

// ---------------- v mean over L per (b,h,d) ----------------
__global__ __launch_bounds__(256) void vmean_kernel(const float* __restrict__ v,
                                                    float* __restrict__ vmean)
{
  __shared__ float red[256];
  int bh = blockIdx.x;
  int h = bh % H_, b = bh / H_;
  int d = threadIdx.x & 63, c = threadIdx.x >> 6;
  float acc = 0.0f;
  for (int l = c * 512; l < (c + 1) * 512; ++l)
    acc += v[((size_t)(b * L_ + l)) * DM_ + h * D_ + d];
  red[threadIdx.x] = acc; __syncthreads();
  if (c == 0)
    vmean[bh * D_ + d] = (red[d] + red[d + 64] + red[d + 128] + red[d + 192]) * (1.0f / L_);
}

// ---------------- broadcast vmean into context (f16) ----------------
__global__ __launch_bounds__(256) void fill_context_kernel(const float* __restrict__ vmean,
                                                           _Float16* __restrict__ ctx_h)
{
  int t = blockIdx.x * 256 + threadIdx.x;            // over B*L*DM
  if (t >= B_ * L_ * DM_) return;
  int c = t % DM_;
  int b = t / (L_ * DM_);
  int h = c / D_, dd = c % D_;
  ctx_h[t] = (_Float16)vmean[(b * H_ + h) * D_ + dd];
}

// ---------------- scatter top-U contexts into context (f16) ----------------
__global__ __launch_bounds__(256) void scatter_ctx_kernel(const float* __restrict__ ctx_top,
                                                          const int* __restrict__ Mtop,
                                                          _Float16* __restrict__ ctx_h)
{
  int t = blockIdx.x * 256 + threadIdx.x;            // over B*H*U*D
  if (t >= B_ * H_ * U_ * D_) return;
  int d  = t % D_;
  int u  = (t / D_) % U_;
  int bh = t / (D_ * U_);
  int b  = bh / H_, h = bh % H_;
  int l  = Mtop[bh * U_ + u];
  ctx_h[((size_t)(b * L_ + l)) * DM_ + h * D_ + d] = (_Float16)ctx_top[t];
}

// ---------------- residual + LayerNorm (one row per block, DM=512) ----------------
template<int STORE_H>
__global__ __launch_bounds__(256) void ln_residual_kernel(
    const float* __restrict__ x, const float* __restrict__ dx,
    const float* __restrict__ g, const float* __restrict__ bta,
    float* __restrict__ out, _Float16* __restrict__ outh)
{
  __shared__ float red[256];
  size_t row = blockIdx.x;
  const float* xp = x + row * DM_;
  const float* dp = dx + row * DM_;
  int t = threadIdx.x;
  float s0 = xp[t] + dp[t];
  float s1 = xp[t + 256] + dp[t + 256];

  red[t] = s0 + s1; __syncthreads();
  for (int s = 128; s > 0; s >>= 1) {
    if (t < s) red[t] += red[t + s];
    __syncthreads();
  }
  float mean = red[0] * (1.0f / DM_); __syncthreads();

  float d0 = s0 - mean, d1 = s1 - mean;
  red[t] = d0 * d0 + d1 * d1; __syncthreads();
  for (int s = 128; s > 0; s >>= 1) {
    if (t < s) red[t] += red[t + s];
    __syncthreads();
  }
  float inv = rsqrtf(red[0] * (1.0f / DM_) + EPS_);

  float o0 = d0 * inv * g[t] + bta[t];
  float o1 = d1 * inv * g[t + 256] + bta[t + 256];
  out[row * DM_ + t] = o0;
  out[row * DM_ + t + 256] = o1;
  if (STORE_H) {
    outh[row * DM_ + t] = (_Float16)o0;
    outh[row * DM_ + t + 256] = (_Float16)o1;
  }
}

// =============================================================================
extern "C" void kernel_launch(void* const* d_in, const int* in_sizes, int n_in,
                              void* d_out, int out_size, void* d_ws, size_t ws_size,
                              hipStream_t stream) {
  const float* x    = (const float*)d_in[0];
  const float* Wq   = (const float*)d_in[1];
  const float* bq   = (const float*)d_in[2];
  const float* Wk   = (const float*)d_in[3];
  const float* bk   = (const float*)d_in[4];
  const float* Wv   = (const float*)d_in[5];
  const float* bv   = (const float*)d_in[6];
  const float* Wo   = (const float*)d_in[7];
  const float* bo   = (const float*)d_in[8];
  const float* Wc1  = (const float*)d_in[9];
  const float* bc1  = (const float*)d_in[10];
  const float* Wc2  = (const float*)d_in[11];
  const float* bc2  = (const float*)d_in[12];
  const float* ln1g = (const float*)d_in[13];
  const float* ln1b = (const float*)d_in[14];
  const float* ln2g = (const float*)d_in[15];
  const float* ln2b = (const float*)d_in[16];
  const int*   sidx = (const int*)d_in[17];
  float* out = (float*)d_out;

  // ---- workspace carve-out ----
  char* p = (char*)d_ws;
  auto alloc = [&](size_t bytes) -> void* {
    void* r = (void*)p;
    p += (bytes + 255) & ~(size_t)255;
    return r;
  };
  const size_t nRows = (size_t)B_ * L_;             // 8192
  _Float16* x_h    = (_Float16*)alloc(nRows * DM_ * 2);
  _Float16* WqT_h  = (_Float16*)alloc((size_t)DM_ * DM_ * 2);
  _Float16* WkT_h  = (_Float16*)alloc((size_t)DM_ * DM_ * 2);
  _Float16* WvT_h  = (_Float16*)alloc((size_t)DM_ * DM_ * 2);
  _Float16* WoT_h  = (_Float16*)alloc((size_t)DM_ * DM_ * 2);
  _Float16* Wc1_h  = (_Float16*)alloc((size_t)DFF_ * DM_ * 2); // already (N,K)
  _Float16* Wc2_h  = (_Float16*)alloc((size_t)DM_ * DFF_ * 2); // already (N,K)
  float* qf        = (float*)alloc(nRows * DM_ * 4);
  float* kf        = (float*)alloc(nRows * DM_ * 4);
  float* vf        = (float*)alloc(nRows * DM_ * 4);
  float* Mscore    = (float*)alloc((size_t)B_ * H_ * L_ * 4);
  int*   Mtop      = (int*)alloc((size_t)B_ * H_ * U_ * 4);
  float* ctx_top   = (float*)alloc((size_t)B_ * H_ * U_ * D_ * 4);
  float* vmean     = (float*)alloc((size_t)B_ * H_ * D_ * 4);
  _Float16* ctx_h  = (_Float16*)alloc(nRows * DM_ * 2);
  float* attn_out  = (float*)alloc(nRows * DM_ * 4);
  float* x1        = (float*)alloc(nRows * DM_ * 4);
  _Float16* x1_h   = (_Float16*)alloc(nRows * DM_ * 2);
  _Float16* h1_h   = (_Float16*)alloc(nRows * DFF_ * 2);
  float* y2        = (float*)alloc(nRows * DM_ * 4);

  const int nX = (int)(nRows * DM_);

  // ---- stage 0: precision conversion / weight transposes ----
  cvt_f16_kernel<<<(nX + 255) / 256, 256, 0, stream>>>(x, x_h, nX);
  transpose_cvt_kernel<<<(DM_ * DM_ + 255) / 256, 256, 0, stream>>>(Wq, WqT_h, DM_, DM_);
  transpose_cvt_kernel<<<(DM_ * DM_ + 255) / 256, 256, 0, stream>>>(Wk, WkT_h, DM_, DM_);
  transpose_cvt_kernel<<<(DM_ * DM_ + 255) / 256, 256, 0, stream>>>(Wv, WvT_h, DM_, DM_);
  transpose_cvt_kernel<<<(DM_ * DM_ + 255) / 256, 256, 0, stream>>>(Wo, WoT_h, DM_, DM_);
  cvt_f16_kernel<<<(DFF_ * DM_ + 255) / 256, 256, 0, stream>>>(Wc1, Wc1_h, DFF_ * DM_);
  cvt_f16_kernel<<<(DM_ * DFF_ + 255) / 256, 256, 0, stream>>>(Wc2, Wc2_h, DM_ * DFF_);

  // ---- stage 1: QKV projections (WMMA) ----
  {
    dim3 grid(DM_ / 32, (unsigned)(nRows / 64));
    gemm_f16_wmma_kernel<0><<<grid, 128, 0, stream>>>(x_h, WqT_h, bq, qf, (_Float16*)nullptr,
                                                      (int)nRows, DM_, DM_);
    gemm_f16_wmma_kernel<0><<<grid, 128, 0, stream>>>(x_h, WkT_h, bk, kf, (_Float16*)nullptr,
                                                      (int)nRows, DM_, DM_);
    gemm_f16_wmma_kernel<0><<<grid, 128, 0, stream>>>(x_h, WvT_h, bv, vf, (_Float16*)nullptr,
                                                      (int)nRows, DM_, DM_);
  }

  // ---- stage 2: ProbSparse selection ----
  sampled_scores_kernel<<<(B_ * H_ * L_) / 256, 256, 0, stream>>>(qf, kf, sidx, Mscore);
  topk_kernel<<<B_ * H_, 256, 0, stream>>>(Mscore, Mtop);
  attn_selected_kernel<<<B_ * H_ * U_, 256, 0, stream>>>(qf, kf, vf, Mtop, ctx_top);

  // ---- stage 3: context assembly ----
  vmean_kernel<<<B_ * H_, 256, 0, stream>>>(vf, vmean);
  fill_context_kernel<<<(nX + 255) / 256, 256, 0, stream>>>(vmean, ctx_h);
  scatter_ctx_kernel<<<(B_ * H_ * U_ * D_ + 255) / 256, 256, 0, stream>>>(ctx_top, Mtop, ctx_h);

  // ---- stage 4: output projection + LN1 ----
  {
    dim3 grid(DM_ / 32, (unsigned)(nRows / 64));
    gemm_f16_wmma_kernel<0><<<grid, 128, 0, stream>>>(ctx_h, WoT_h, bo, attn_out,
                                                      (_Float16*)nullptr, (int)nRows, DM_, DM_);
  }
  ln_residual_kernel<1><<<(unsigned)nRows, 256, 0, stream>>>(x, attn_out, ln1g, ln1b, x1, x1_h);

  // ---- stage 5: FFN (WMMA) + LN2 ----
  {
    dim3 grid1(DFF_ / 32, (unsigned)(nRows / 64));
    gemm_f16_wmma_kernel<1><<<grid1, 128, 0, stream>>>(x1_h, Wc1_h, bc1, (float*)nullptr,
                                                       h1_h, (int)nRows, DFF_, DM_);
    dim3 grid2(DM_ / 32, (unsigned)(nRows / 64));
    gemm_f16_wmma_kernel<0><<<grid2, 128, 0, stream>>>(h1_h, Wc2_h, bc2, y2,
                                                       (_Float16*)nullptr, (int)nRows, DM_, DFF_);
  }
  ln_residual_kernel<0><<<(unsigned)nRows, 256, 0, stream>>>(x1, y2, ln2g, ln2b, out,
                                                             (_Float16*)nullptr);
}
